// PairSelfAttention_60395830116865
// MI455X (gfx1250) — matrix operand order repaired and checked by reading, not compiled, not run on hardware
//
#include <hip/hip_runtime.h>
#include <hip/hip_bf16.h>
#include <math.h>

// ---------------- problem constants ----------------
constexpr int B = 32, S = 512, E = 256, H = 8, D = 32;
constexpr int BHSD = B * H * S * D;

typedef __attribute__((ext_vector_type(16))) _Float16 v16h;
typedef __attribute__((ext_vector_type(8)))  _Float16 v8h;
typedef __attribute__((ext_vector_type(4)))  _Float16 v4h;
typedef __attribute__((ext_vector_type(8)))  float    v8f;
typedef __attribute__((ext_vector_type(4)))  float    v4f;

// v16h fragment assembled from two 16-byte halves (elements 0-7 / 8-15)
union Frag {
    v16h f;
    struct { v8h lo, hi; } p;
};

// CDNA5 wave32 16x16x32 f16 WMMA layouts (CDNA5 ISA 7.12.2):
//  A (16x32): lane half h=lane>>4, row M=lane&15;
//             elems 0-7 -> K=8h+e (contig), elems 8-15 -> K=16+8h+(e-8) (contig)
//  B (32x16): col N=lane&15; this code uses elems 0-15 -> K=16h+e (contig) for A and B alike
//  C/D (16x16): elem r -> M=8h+r, N=lane&15

__device__ __forceinline__ v8f wmma_f16(const v16h& a, const v16h& b, const v8f& c) {
    return __builtin_amdgcn_wmma_f32_16x16x32_f16(
        false, a, false, b, (short)0, c, false, false);
}

// A-fragment from a row-major (rows x 32) f16 matrix, given row pointer
__device__ __forceinline__ v16h load_a_frag(const _Float16* rowp, int half) {
    Frag u;
    u.p.lo = *(const v8h*)(rowp + 16 * half);
    u.p.hi = *(const v8h*)(rowp + 16 * half + 8);
    return u.f;
}
// B-fragment where B[k][n] = colsrc_n[k], colsrc_n = 32 contiguous halves
__device__ __forceinline__ v16h load_b_frag(const _Float16* colp, int half) {
    Frag u;
    u.p.lo = *(const v8h*)(colp + 16 * half);
    u.p.hi = *(const v8h*)(colp + 16 * half + 8);
    return u.f;
}

// ============================================================
// Kernel 0: transpose + f16-convert the three 32x32 weights
//   Wt[n*32 + k] = (f16) W[k*32 + n]
// ============================================================
__global__ void wcvt_kernel(const float* __restrict__ Wq,
                            const float* __restrict__ Wk,
                            const float* __restrict__ Wv,
                            _Float16* __restrict__ wqt,
                            _Float16* __restrict__ wkt,
                            _Float16* __restrict__ wvt) {
    const int idx = threadIdx.x + blockIdx.x * blockDim.x;   // 0..1023
    if (idx >= D * D) return;
    const int n = idx >> 5, k = idx & 31;
    wqt[idx] = (_Float16)Wq[k * D + n];
    wkt[idx] = (_Float16)Wk[k * D + n];
    wvt[idx] = (_Float16)Wv[k * D + n];
}

// ============================================================
// Kernel 1: pair selection (IoU argmax + L1 orientation)
// ============================================================
__global__ void pairs_kernel(const float* __restrict__ c,
                             int* __restrict__ p0, int* __restrict__ p1) {
    __shared__ float sx0[S], sy0[S], sx1[S], sy1[S], sar[S], sl1[S];
    const int b = blockIdx.x;
    const float* cb = c + (size_t)b * S * 4;

    for (int i = threadIdx.x; i < S; i += blockDim.x) {
        v4f cv = *(const v4f*)(cb + i * 4);
        float x0 = cv.x - 0.5f * cv.w, y0 = cv.y - 0.5f * cv.z;
        float x1 = cv.x + 0.5f * cv.w, y1 = cv.y + 0.5f * cv.z;
        sx0[i] = x0; sy0[i] = y0; sx1[i] = x1; sy1[i] = y1;
        sar[i] = (x1 - x0) * (y1 - y0);
        sl1[i] = cv.z - cv.x + (cv.w - cv.y);
    }
    __syncthreads();

    for (int i = threadIdx.x; i < S; i += blockDim.x) {
        const float ax0 = sx0[i], ay0 = sy0[i], ax1 = sx1[i], ay1 = sy1[i], aar = sar[i];
        float best = -3.4e38f;
        int   bidx = 0;
        for (int j = 0; j < S; ++j) {
            float w  = fmaxf(fminf(ax1, sx1[j]) - fmaxf(ax0, sx0[j]), 0.0f);
            float hh = fmaxf(fminf(ay1, sy1[j]) - fmaxf(ay0, sy0[j]), 0.0f);
            float inter = w * hh;
            float iou = inter / (aar + sar[j] - inter);
            if (j == i) iou -= 1.0f;                    // - eye
            if (iou > best) { best = iou; bidx = j; }   // first-max, as jnp.argmax
        }
        bool cond = (sl1[i] >= sl1[bidx]);
        p0[b * S + i] = cond ? i : bidx;
        p1[b * S + i] = cond ? bidx : i;
    }
}

// ============================================================
// Kernel 2: QKV projections via WMMA, all stores contiguous 16B
//   q,k row-major (S x D): computed TRANSPOSED as Wt(A) x X^T(B)
//   v transposed (D x S):  computed as X(A) x Wv(B)
// ============================================================
__global__ void proj_kernel(const float* __restrict__ content,
                            const float* __restrict__ pos,
                            const _Float16* __restrict__ wqt,
                            const _Float16* __restrict__ wkt,
                            const _Float16* __restrict__ wvt,
                            _Float16* __restrict__ qh,
                            _Float16* __restrict__ kh,
                            _Float16* __restrict__ vth) {
    const int lane = threadIdx.x;
    const int half = lane >> 4;
    const int mrow = lane & 15;

    const int tiles = S / 16;
    const int it = blockIdx.x % tiles;
    const int h  = (blockIdx.x / tiles) % H;
    const int b  = blockIdx.x / (tiles * H);
    const int s_base = it * 16;

    const size_t row0 = ((size_t)b * S + s_base + mrow) * E + h * D;

    // x_qk in B-fragment order (k = 16h+e, contiguous 16 floats)
    v4f cb0 = *(const v4f*)(content + row0 + 16 * half);
    v4f cb1 = *(const v4f*)(content + row0 + 16 * half + 4);
    v4f cb2 = *(const v4f*)(content + row0 + 16 * half + 8);
    v4f cb3 = *(const v4f*)(content + row0 + 16 * half + 12);
    v4f pb0 = *(const v4f*)(pos + row0 + 16 * half);
    v4f pb1 = *(const v4f*)(pos + row0 + 16 * half + 4);
    v4f pb2 = *(const v4f*)(pos + row0 + 16 * half + 8);
    v4f pb3 = *(const v4f*)(pos + row0 + 16 * half + 12);
    v16h xb;
#pragma unroll
    for (int e = 0; e < 4; ++e) {
        xb[e]      = (_Float16)(cb0[e] + pb0[e]);
        xb[e + 4]  = (_Float16)(cb1[e] + pb1[e]);
        xb[e + 8]  = (_Float16)(cb2[e] + pb2[e]);
        xb[e + 12] = (_Float16)(cb3[e] + pb3[e]);
    }

    // x_v in A-fragment order (k = 16h+e, contiguous)
    v4f ca0 = *(const v4f*)(content + row0 + 16 * half);
    v4f ca1 = *(const v4f*)(content + row0 + 16 * half + 4);
    v4f ca2 = *(const v4f*)(content + row0 + 16 * half + 8);
    v4f ca3 = *(const v4f*)(content + row0 + 16 * half + 12);
    v16h xv;
#pragma unroll
    for (int e = 0; e < 4; ++e) {
        xv[e]      = (_Float16)ca0[e];
        xv[e + 4]  = (_Float16)ca1[e];
        xv[e + 8]  = (_Float16)ca2[e];
        xv[e + 12] = (_Float16)ca3[e];
    }

    const size_t obase = (((size_t)b * H + h) * S) * D;

    // ---- q, k: C' = Wt(A) x X^T(B); elem r -> n = n_base+8h+r (contig) ----
    const _Float16* Wts[2] = { wqt, wkt };
    _Float16*       dsts[2] = { qh, kh };
#pragma unroll
    for (int w = 0; w < 2; ++w) {
#pragma unroll
        for (int nt = 0; nt < 2; ++nt) {
            const int n_base = nt * 16;
            const v16h af = load_a_frag(Wts[w] + (size_t)(n_base + mrow) * D, half);
            v8f acc = {};
            acc = wmma_f16(af, xb, acc);
            v8h o;
#pragma unroll
            for (int r = 0; r < 8; ++r) o[r] = (_Float16)acc[r];
            *(v8h*)(dsts[w] + obase + (size_t)(s_base + mrow) * D + n_base + 8 * half) = o;
        }
    }

    // ---- v: C = X(A) x Wv(B); elem r -> s = s_base+8h+r (contig in DxS) ----
#pragma unroll
    for (int nt = 0; nt < 2; ++nt) {
        const int n_base = nt * 16;
        const v16h bf = load_b_frag(wvt + (size_t)(n_base + mrow) * D, half);
        v8f acc = {};
        acc = wmma_f16(xv, bf, acc);
        v8h o;
#pragma unroll
        for (int r = 0; r < 8; ++r) o[r] = (_Float16)acc[r];
        *(v8h*)(vth + obase + (size_t)(n_base + mrow) * S + s_base + 8 * half) = o;
    }
}

// ============================================================
// Kernel 3: fused pair attention (one wave per 16-row tile)
// ============================================================
__global__ void attn_kernel(const _Float16* __restrict__ qh,
                            const _Float16* __restrict__ kh,
                            const _Float16* __restrict__ vth,
                            const int* __restrict__ p0,
                            const int* __restrict__ p1,
                            float* __restrict__ out) {
    __shared__ float    lds_logits[16 * S]; // 32 KB (reused as out-stage)
    __shared__ _Float16 lds_attn[16 * S];   // 16 KB

    const int lane = threadIdx.x;
    const int half = lane >> 4;
    const int mrow = lane & 15;

    const int tiles = S / 16;
    const int it = blockIdx.x % tiles;
    const int h  = (blockIdx.x / tiles) % H;
    const int b  = blockIdx.x / (tiles * H);
    const int i_base = it * 16;

    const size_t hb = (((size_t)b * H + h) * S) * D;
    const int*   p0b = p0 + b * S;
    const int*   p1b = p1 + b * S;
    const float  inv_scale = rsqrtf(2.0f * (float)E);   // 1/sqrt(512)

    // ---- Q fragments (rows gathered via p0/p1), 2 x b128 each ----
    const int r0i = p0b[i_base + mrow];
    const int r1i = p1b[i_base + mrow];
    const v16h q0f = load_a_frag(qh + hb + (size_t)r0i * D, half);
    const v16h q1f = load_a_frag(qh + hb + (size_t)r1i * D, half);

    // ---- Phase 1: logits = (Q0 K0^T + Q1 K1^T) * inv_scale -> LDS ----
    for (int jt = 0; jt < S / 16; ++jt) {
        const int j_base = jt * 16;
        const int rk0 = p0b[j_base + mrow];
        const int rk1 = p1b[j_base + mrow];
        const v16h k0f = load_b_frag(kh + hb + (size_t)rk0 * D, half);
        const v16h k1f = load_b_frag(kh + hb + (size_t)rk1 * D, half);
        v8f acc = {};
        acc = wmma_f16(q0f, k0f, acc);
        acc = wmma_f16(q1f, k1f, acc);
#pragma unroll
        for (int r = 0; r < 8; ++r)
            lds_logits[(8 * half + r) * S + j_base + mrow] = acc[r] * inv_scale;
    }
    __syncthreads();

    // ---- Phase 2: softmax over j (each lane scans half of its row) ----
    {
        const float* rowp = lds_logits + mrow * S + half * (S / 2);
        float m = -3.4e38f;
        for (int j = 0; j < S / 8; ++j) {
            v4f x = *(const v4f*)(rowp + j * 4);
            m = fmaxf(m, fmaxf(fmaxf(x.x, x.y), fmaxf(x.z, x.w)));
        }
        m = fmaxf(m, __shfl_xor(m, 16, 32));
        float s = 0.0f;
        for (int j = 0; j < S / 8; ++j) {
            v4f x = *(const v4f*)(rowp + j * 4);
            s += expf(x.x - m) + expf(x.y - m) + expf(x.z - m) + expf(x.w - m);
        }
        s += __shfl_xor(s, 16, 32);
        const float inv = 1.0f / s;
        _Float16* arow = lds_attn + mrow * S + half * (S / 2);
        for (int j = 0; j < S / 8; ++j) {
            v4f x = *(const v4f*)(rowp + j * 4);
            v4h pr;
            pr[0] = (_Float16)(expf(x.x - m) * inv);
            pr[1] = (_Float16)(expf(x.y - m) * inv);
            pr[2] = (_Float16)(expf(x.z - m) * inv);
            pr[3] = (_Float16)(expf(x.w - m) * inv);
            *(v4h*)(arow + j * 4) = pr;
        }
    }
    __syncthreads();

    // ---- Phase 3: O = attn @ V (V transposed: D x S, contiguous in k) ----
    float* lds_out = lds_logits;  // 16 x 32 fp32 stage (logits are dead now)
#pragma unroll
    for (int dt = 0; dt < 2; ++dt) {
        const int d_base = dt * 16;
        const _Float16* vcol = vth + hb + (size_t)(d_base + mrow) * S;
        v8f acc = {};
        for (int kt = 0; kt < S / 32; ++kt) {
            const int jb = kt * 32;
            const v16h af = load_a_frag(lds_attn + mrow * S + jb, half);
            const v16h bf = load_b_frag(vcol + jb, half);
            acc = wmma_f16(af, bf, acc);
        }
#pragma unroll
        for (int r = 0; r < 8; ++r)
            lds_out[(8 * half + r) * 32 + d_base + mrow] = acc[r];
    }
    __syncthreads();

    // contiguous 4 x b128 store per lane: row mrow, floats 16*half..16*half+15
    {
        const float* orow = lds_out + mrow * 32 + 16 * half;
        float* optr = out + ((size_t)b * S + i_base + mrow) * E + h * D + 16 * half;
#pragma unroll
        for (int c = 0; c < 4; ++c)
            *(v4f*)(optr + 4 * c) = *(const v4f*)(orow + 4 * c);
    }
}

// ============================================================
// Launcher
// ============================================================
extern "C" void kernel_launch(void* const* d_in, const int* in_sizes, int n_in,
                              void* d_out, int out_size, void* d_ws, size_t ws_size,
                              hipStream_t stream) {
    (void)in_sizes; (void)n_in; (void)out_size; (void)ws_size;

    const float* content = (const float*)d_in[0];
    const float* pos     = (const float*)d_in[1];
    const float* centers = (const float*)d_in[2];
    const float* Wq      = (const float*)d_in[3];
    const float* Wk      = (const float*)d_in[4];
    const float* Wv      = (const float*)d_in[5];
    float* out = (float*)d_out;

    // workspace: p0, p1 (int) | qh, kh (SxD f16) | vth (DxS f16) | Wt f16 x3
    int* p0 = (int*)d_ws;
    int* p1 = p0 + B * S;
    _Float16* qh  = (_Float16*)(p1 + B * S);
    _Float16* kh  = qh + BHSD;
    _Float16* vth = kh + BHSD;
    _Float16* wqt = vth + BHSD;
    _Float16* wkt = wqt + D * D;
    _Float16* wvt = wkt + D * D;

    wcvt_kernel<<<4, 256, 0, stream>>>(Wq, Wk, Wv, wqt, wkt, wvt);
    pairs_kernel<<<B, 256, 0, stream>>>(centers, p0, p1);

    const int nblocks = B * H * (S / 16); // 8192 one-wave workgroups
    proj_kernel<<<nblocks, 32, 0, stream>>>(content, pos, wqt, wkt, wvt, qh, kh, vth);
    attn_kernel<<<nblocks, 32, 0, stream>>>(qh, kh, vth, p0, p1, out);
}
